// MultiHeadAttention_10943576670404
// MI455X (gfx1250) — compile-verified, matrix operand order checked by reading, not run on hardware
//
#include <hip/hip_runtime.h>

// ---------------- problem constants ----------------
#define NHEADS 16
#define HDIM   64
#define CEMB   1024        // NHEADS*HDIM
#define BB     4
#define TT     2048
#define BT     (BB * TT)   // 8192 rows
#define C3     (3 * CEMB)  // 3072

#define LDSROW 40          // padded LDS row pitch in bf16 (80B = 5*16B, bank-conflict free)

typedef __attribute__((ext_vector_type(16))) __bf16 bf16x16;
typedef __attribute__((ext_vector_type(8)))  __bf16 bf16x8;
typedef __attribute__((ext_vector_type(8)))  float  floatx8;
typedef __attribute__((ext_vector_type(4)))  unsigned uint32x4;
typedef __attribute__((ext_vector_type(8)))  int      int32x8;
typedef __attribute__((ext_vector_type(4)))  int      int32x4;

// ---------------- WMMA wrapper ----------------
__device__ __forceinline__ floatx8 wmma_bf16(bf16x16 a, bf16x16 b, floatx8 c) {
  // D(16x16,f32) = A(16x32,bf16) x B(32x16,bf16) + C
  return __builtin_amdgcn_wmma_f32_16x16x32_bf16(
      /*neg_a=*/false, a, /*neg_b=*/false, b,
      /*c_mod=*/(short)0, c, /*reuse_a=*/false, /*reuse_b=*/false);
}

// Load a 16x32 bf16 fragment from a row-major [16][ld] tile in GLOBAL memory.
// A-matrix layout (16-bit, 16x32): lanes 0-15 row M=lane, K: lo=0..7 hi=16..23;
// lanes 16-31 row M=lane-16, K: lo=8..15 hi=24..31.  Same pattern serves the
// B operand when the matrix is stored [N][K] (i.e. B^T row-major).
__device__ __forceinline__ bf16x16 load_frag(const __bf16* __restrict__ base,
                                             int ld, int lane) {
  const int r  = lane & 15;
  const int hf = lane >> 4;
  const __bf16* p = base + (size_t)r * ld + hf * 8;
  bf16x8 lo = *(const bf16x8*)(p);        // K = 8*hf .. 8*hf+7
  bf16x8 hi = *(const bf16x8*)(p + 16);   // K = 16+8*hf .. 16+8*hf+7
  bf16x16 f;
#pragma unroll
  for (int i = 0; i < 8; ++i) { f[i] = lo[i]; f[i + 8] = hi[i]; }
  return f;
}

// Same fragment pattern from an LDS-staged panel with LDSROW pitch.
__device__ __forceinline__ bf16x16 lds_frag(const __bf16* base, int lane) {
  const int r  = lane & 15;
  const int hf = lane >> 4;
  const __bf16* p = base + r * LDSROW + hf * 8;
  bf16x8 lo = *(const bf16x8*)(p);        // ds_load_b128, 16B aligned (pitch=80B)
  bf16x8 hi = *(const bf16x8*)(p + 16);
  bf16x16 f;
#pragma unroll
  for (int i = 0; i < 8; ++i) { f[i] = lo[i]; f[i + 8] = hi[i]; }
  return f;
}

// ---------------- Tensor Data Mover: 2D tile (32 x 128 rows) -> LDS ----------
// Builds the D# descriptor per CDNA5 ISA sec 8.3-8.6 and issues
// tensor_load_to_lds (TENSORcnt-tracked async DMA).  LDS padding: 4 DWORDs
// after every 16 DWORDs => 80B row pitch (matches LDSROW).
__device__ __forceinline__ void tdm_load_panel(unsigned lds_byte_addr,
                                               const __bf16* gptr,
                                               unsigned row_stride_elems,
                                               unsigned tensor_rows) {
  const unsigned long long ga = (unsigned long long)(size_t)gptr;
  uint32x4 g0;
  g0[0] = 1u;                                        // count=1, user mode, no gather
  g0[1] = lds_byte_addr;                             // LDS dest (bytes)
  g0[2] = (unsigned)(ga & 0xFFFFFFFFu);              // global_addr[31:0]
  g0[3] = (unsigned)((ga >> 32) & 0x01FFFFFFu) | (2u << 30);  // addr[56:32] | type=2
  const unsigned td0 = row_stride_elems;             // tensor width (elems)
  const unsigned td1 = tensor_rows;                  // tensor height (rows)
  int32x8 g1;
  g1[0] = (int)((1u << 16)       // data_size = 2 bytes
              | (1u << 20)       // pad_enable
              | (3u << 22)       // pad_interval: 16 DWORDs (64B, one tile row)
              | (3u << 25));     // pad_amount: 4 DWORDs (16B)
  g1[1] = (int)((td0 & 0xFFFFu) << 16);                              // tensor_dim0 lo
  g1[2] = (int)(((td0 >> 16) & 0xFFFFu) | ((td1 & 0xFFFFu) << 16));  // td0 hi | td1 lo
  g1[3] = (int)(((td1 >> 16) & 0xFFFFu) | (32u << 16));              // td1 hi | tile_dim0=32
  g1[4] = (int)(128u);                               // tile_dim1=128, tile_dim2=0
  g1[5] = (int)row_stride_elems;                     // tensor_dim0_stride[31:0]
  g1[6] = 0;                                         // stride hi | dim1_stride lo
  g1[7] = 0;
  const int32x4 gz = {0, 0, 0, 0};                   // groups 2/3: dims>2 unused
  asm volatile("tensor_load_to_lds %0, %1, %2, %3"
               :
               : "s"(g0), "s"(g1), "s"(gz), "s"(gz)
               : "memory");
}

// ---------------- elementwise converters ----------------
__global__ void cvt_f32_bf16(const float* __restrict__ s, __bf16* __restrict__ d,
                             size_t n) {
  size_t i = (size_t)blockIdx.x * blockDim.x + threadIdx.x;
  if (i < n) d[i] = (__bf16)s[i];
}

// src [K][N] f32  ->  dst [N][K] bf16   (weights stored transposed for B-frags)
__global__ void cvt_transpose(const float* __restrict__ s, __bf16* __restrict__ d,
                              int K, int N) {
  size_t i = (size_t)blockIdx.x * blockDim.x + threadIdx.x;
  if (i >= (size_t)K * N) return;
  int n = (int)(i % N);
  int k = (int)(i / N);
  d[(size_t)n * K + k] = (__bf16)s[i];
}

// qkv [BT][3C] bf16 -> q,k [B][NH][T][HD] ; v transposed [B][NH][HD][T]
__global__ void split_heads(const __bf16* __restrict__ qkv,
                            __bf16* __restrict__ q, __bf16* __restrict__ k,
                            __bf16* __restrict__ vT) {
  size_t i = (size_t)blockIdx.x * blockDim.x + threadIdx.x;
  if (i >= (size_t)BT * C3) return;
  int col = (int)(i % C3);
  size_t bt = i / C3;
  int t = (int)(bt % TT);
  int b = (int)(bt / TT);
  __bf16 val = qkv[i];
  int which = col / CEMB;
  int c = col % CEMB;
  int head = c / HDIM, h = c % HDIM;
  size_t hb = (size_t)(b * NHEADS + head);
  if (which == 0)      q[(hb * TT + t) * HDIM + h] = val;
  else if (which == 1) k[(hb * TT + t) * HDIM + h] = val;
  else                vT[(hb * HDIM + h) * TT + t] = val;
}

// ---------------- WMMA GEMM with TDM-staged LDS panels ----------------------
// C[M][N] = A[M][K] * BT[N][K]^T.  256 threads = 8 waves as 2(M)x4(N); wave
// tile 64Mx32N (4 A-frags x 2 B-frags -> 8 WMMAs per k-step).  Block tile
// 128x128.  Wave 0 drives tensor_load_to_lds DMA of the 128x32 A and B
// k-panels into double-buffered LDS; compute waves read conflict-free
// ds_load_b128 fragments.
template <bool F32OUT>
__global__ void __launch_bounds__(256)
gemm_wmma(const __bf16* __restrict__ A, const __bf16* __restrict__ BT_,
          const float* __restrict__ bias, void* __restrict__ Cout,
          int M, int N, int K) {
  __shared__ alignas(16) __bf16 Alds[2][128 * LDSROW];
  __shared__ alignas(16) __bf16 Blds[2][128 * LDSROW];

  const int lane  = threadIdx.x & 31;
  const int w     = threadIdx.x >> 5;
  const int mblk  = blockIdx.y * 128;
  const int nblk  = blockIdx.x * 128;
  const int wmoff = (w >> 2) * 64;
  const int wnoff = (w & 3) * 32;

  // first panel
  if (w == 0) {
    tdm_load_panel((unsigned)(size_t)&Alds[0][0], A + (size_t)mblk * K, K, M - mblk);
    tdm_load_panel((unsigned)(size_t)&Blds[0][0], BT_ + (size_t)nblk * K, K, N - nblk);
    __builtin_amdgcn_s_wait_tensorcnt(0);
  }
  __syncthreads();

  floatx8 acc[4][2] = {};
  for (int k0 = 0, it = 0; k0 < K; k0 += 32, ++it) {
    const int buf = it & 1;
    const bool more = (k0 + 32) < K;
    if (w == 0 && more) {   // overlap next panel DMA with this panel's compute
      tdm_load_panel((unsigned)(size_t)&Alds[buf ^ 1][0],
                     A + (size_t)mblk * K + (k0 + 32), K, M - mblk);
      tdm_load_panel((unsigned)(size_t)&Blds[buf ^ 1][0],
                     BT_ + (size_t)nblk * K + (k0 + 32), K, N - nblk);
    }
    bf16x16 af[4], bf[2];
#pragma unroll
    for (int i = 0; i < 4; ++i)
      af[i] = lds_frag(&Alds[buf][(wmoff + 16 * i) * LDSROW], lane);
#pragma unroll
    for (int j = 0; j < 2; ++j)
      bf[j] = lds_frag(&Blds[buf][(wnoff + 16 * j) * LDSROW], lane);
#pragma unroll
    for (int i = 0; i < 4; ++i)
#pragma unroll
      for (int j = 0; j < 2; ++j)
        acc[i][j] = wmma_bf16(af[i], bf[j], acc[i][j]);
    if (w == 0 && more) __builtin_amdgcn_s_wait_tensorcnt(0);
    __syncthreads();
  }

  const int r = lane & 15, hf = lane >> 4;
  const int m = mblk + wmoff, n = nblk + wnoff;
#pragma unroll
  for (int i = 0; i < 4; ++i)
#pragma unroll
    for (int j = 0; j < 2; ++j) {
      const int nn = n + 16 * j + r;
#pragma unroll
      for (int e = 0; e < 8; ++e) {
        const size_t idx = (size_t)(m + 16 * i + e + 8 * hf) * N + nn;
        if constexpr (F32OUT) ((float*)Cout)[idx]  = acc[i][j][e] + bias[nn];
        else                  ((__bf16*)Cout)[idx] = (__bf16)acc[i][j][e];
      }
    }
}

// ---------------- causal flash attention, one wave per 16-row Q tile --------
__global__ void __launch_bounds__(32)
flash_attn(const __bf16* __restrict__ Qh, const __bf16* __restrict__ Kh,
           const __bf16* __restrict__ VT, __bf16* __restrict__ Y) {
  const int MT   = TT / 16;
  const int wg   = blockIdx.x;
  const int mt   = wg % MT;
  const int head = (wg / MT) % NHEADS;
  const int b    = wg / (MT * NHEADS);
  const int lane = threadIdx.x;
  const int r    = lane & 15;
  const int hf   = lane >> 4;

  const size_t hb = (size_t)(b * NHEADS + head);
  const __bf16* q  = Qh + hb * TT * HDIM;
  const __bf16* k  = Kh + hb * TT * HDIM;
  const __bf16* vt = VT + hb * (size_t)HDIM * TT;
  const int m0 = mt * 16;

  // Q fragments for K-dim 0..31 and 32..63
  const bf16x16 qa0 = load_frag(q + (size_t)m0 * HDIM, HDIM, lane);
  const bf16x16 qa1 = load_frag(q + (size_t)m0 * HDIM + 32, HDIM, lane);

  float mrow[8], lrow[8];
  floatx8 o[4] = {};
#pragma unroll
  for (int i = 0; i < 8; ++i) { mrow[i] = -1e30f; lrow[i] = 0.0f; }

  __shared__ alignas(16) __bf16 Plds[16 * 32];   // P tile, C-layout -> A-layout

  const float SCALE = 0.125f;                    // 1/sqrt(64)
  const int s_end = m0 + 16;                     // exclusive causal bound
  for (int s0 = 0; s0 < s_end; s0 += 32) {
    floatx8 sacc0 = {}, sacc1 = {};
    {
      bf16x16 kb0 = load_frag(k + (size_t)s0 * HDIM, HDIM, lane);
      bf16x16 kb1 = load_frag(k + (size_t)s0 * HDIM + 32, HDIM, lane);
      sacc0 = wmma_bf16(qa0, kb0, sacc0);
      sacc0 = wmma_bf16(qa1, kb1, sacc0);
    }
    if (s0 + 16 < s_end) {
      bf16x16 kb0 = load_frag(k + (size_t)(s0 + 16) * HDIM, HDIM, lane);
      bf16x16 kb1 = load_frag(k + (size_t)(s0 + 16) * HDIM + 32, HDIM, lane);
      sacc1 = wmma_bf16(qa0, kb0, sacc1);
      sacc1 = wmma_bf16(qa1, kb1, sacc1);
    }
    // online softmax; C-frag element i = (row m0+i+8*hf, col s0[+16]+r)
#pragma unroll
    for (int i = 0; i < 8; ++i) {
      const int row = m0 + i + 8 * hf;
      float x0 = (s0 + r <= row)      ? sacc0[i] * SCALE : -1e30f;
      float x1 = (s0 + 16 + r <= row) ? sacc1[i] * SCALE : -1e30f;
      float tm = fmaxf(x0, x1);
#pragma unroll
      for (int off = 1; off < 16; off <<= 1)
        tm = fmaxf(tm, __shfl_xor(tm, off, 32));
      const float mn   = fmaxf(mrow[i], tm);
      const float corr = __expf(mrow[i] - mn);
      const float p0   = __expf(x0 - mn);
      const float p1   = __expf(x1 - mn);
      float rs = p0 + p1;
#pragma unroll
      for (int off = 1; off < 16; off <<= 1)
        rs += __shfl_xor(rs, off, 32);
      lrow[i] = lrow[i] * corr + rs;
      mrow[i] = mn;
      o[0][i] *= corr; o[1][i] *= corr; o[2][i] *= corr; o[3][i] *= corr;
      const int lr = i + 8 * hf;
      Plds[lr * 32 + r]      = (__bf16)p0;
      Plds[lr * 32 + 16 + r] = (__bf16)p1;
    }
    __syncthreads();
    // re-load P in A-fragment layout
    bf16x16 pf;
    {
      const __bf16* pp = &Plds[r * 32 + hf * 8];
      bf16x8 lo = *(const bf16x8*)(pp);
      bf16x8 hi = *(const bf16x8*)(pp + 16);
#pragma unroll
      for (int i = 0; i < 8; ++i) { pf[i] = lo[i]; pf[i + 8] = hi[i]; }
    }
#pragma unroll
    for (int ht = 0; ht < 4; ++ht) {
      bf16x16 vf = load_frag(vt + (size_t)(ht * 16) * TT + s0, TT, lane);
      o[ht] = wmma_bf16(pf, vf, o[ht]);
    }
    __syncthreads();
  }

  // epilogue: normalize and scatter into y [B][T][NH*HD] (bf16 for proj GEMM)
#pragma unroll
  for (int i = 0; i < 8; ++i) {
    const float inv = 1.0f / lrow[i];
    const int t = m0 + i + 8 * hf;
    const size_t base = ((size_t)b * TT + t) * CEMB + head * HDIM;
#pragma unroll
    for (int ht = 0; ht < 4; ++ht)
      Y[base + ht * 16 + r] = (__bf16)(o[ht][i] * inv);
  }
}

// ---------------- host launcher ----------------
extern "C" void kernel_launch(void* const* d_in, const int* in_sizes, int n_in,
                              void* d_out, int out_size, void* d_ws, size_t ws_size,
                              hipStream_t stream) {
  (void)in_sizes; (void)n_in; (void)out_size; (void)ws_size;
  const float* x      = (const float*)d_in[0];   // [B,T,C]
  const float* W_attn = (const float*)d_in[1];   // [C, 3C]
  const float* W_proj = (const float*)d_in[2];   // [C, C]
  const float* b_proj = (const float*)d_in[3];   // [C]
  float* out = (float*)d_out;                    // [B,T,C] f32

  char* ws = (char*)d_ws;
  size_t off = 0;
  auto alloc = [&](size_t bytes) -> void* {
    off = (off + 255) & ~(size_t)255;
    void* p = ws + off;
    off += bytes;
    return p;
  };
  __bf16* xb   = (__bf16*)alloc((size_t)BT * CEMB * 2);        // 16 MB
  __bf16* WaT  = (__bf16*)alloc((size_t)CEMB * C3 * 2);        //  6 MB
  __bf16* WpT  = (__bf16*)alloc((size_t)CEMB * CEMB * 2);      //  2 MB
  __bf16* qkv  = (__bf16*)alloc((size_t)BT * C3 * 2);          // 48 MB
  __bf16* qh   = (__bf16*)alloc((size_t)BT * CEMB * 2);        // 16 MB
  __bf16* kh   = (__bf16*)alloc((size_t)BT * CEMB * 2);        // 16 MB
  __bf16* vT   = (__bf16*)alloc((size_t)BT * CEMB * 2 + 256);  // 16 MB (+pad)
  __bf16* y    = qkv;  // qkv is dead after split_heads; reuse for attention out

  const int THR = 256;
  // 1) precision conversion / weight transposition
  {
    size_t n = (size_t)BT * CEMB;
    cvt_f32_bf16<<<(unsigned)((n + THR - 1) / THR), THR, 0, stream>>>(x, xb, n);
  }
  {
    size_t n = (size_t)CEMB * C3;
    cvt_transpose<<<(unsigned)((n + THR - 1) / THR), THR, 0, stream>>>(W_attn, WaT, CEMB, C3);
  }
  {
    size_t n = (size_t)CEMB * CEMB;
    cvt_transpose<<<(unsigned)((n + THR - 1) / THR), THR, 0, stream>>>(W_proj, WpT, CEMB, CEMB);
  }
  // 2) qkv = x @ W_attn        (WMMA + TDM-staged LDS, 64x32 wave tiles)
  gemm_wmma<false><<<dim3(C3 / 128, BT / 128), THR, 0, stream>>>(
      xb, WaT, nullptr, qkv, BT, C3, CEMB);
  // 3) head split (+ v transpose)
  {
    size_t n = (size_t)BT * C3;
    split_heads<<<(unsigned)((n + THR - 1) / THR), THR, 0, stream>>>(qkv, qh, kh, vT);
  }
  // 4) causal flash attention  (WMMA)
  flash_attn<<<BB * NHEADS * (TT / 16), 32, 0, stream>>>(qh, kh, vT, y);
  // 5) out = y @ W_proj + b    (WMMA + TDM, f32 out)
  gemm_wmma<true><<<dim3(CEMB / 128, BT / 128), THR, 0, stream>>>(
      y, WpT, b_proj, out, BT, CEMB, CEMB);
}
